// PPO_89593017795215
// MI455X (gfx1250) — matrix operand order repaired and checked by reading, not compiled, not run on hardware
//
#include <hip/hip_runtime.h>
#include <math.h>

// ---------------- constants ----------------
#define GAMMA_F   0.99f
#define CDEC      0.9405f                 // GAMMA * LAMBDA
#define HALF_L2PI 0.9189385332046727f     // 0.5*log(2*pi)

constexpr int T_N = 262144;
constexpr int D_N = 128;    // state dim (K of first GEMM)
constexpr int H_N = 256;    // hidden    (N of first GEMM)
constexpr int A_N = 8;

constexpr int WAVES        = 4;     // waves per block in forward kernel
constexpr int ROWS_PER_BLK = WAVES * 16;   // 64
constexpr int KP           = 132;   // LDS pitch (bf16 elems) for transposed weights

constexpr int SCAN_L  = 512;              // elements per scan block
constexpr int SCAN_NB = T_N / SCAN_L;     // 512 scan blocks
constexpr int ACT_NB  = T_N / 256;        // 1024 actor partial blocks

typedef __attribute__((ext_vector_type(16))) __bf16 v16bf;
typedef __attribute__((ext_vector_type(8)))  float  v8f;

// Branch-free tanh: sign(x)*(1-e)/(1+e), e = exp(-2|x|).
// Uses hw v_exp_f32 + v_rcp_f32; no EXEC divergence (WMMA loops stay all-lanes).
__device__ __forceinline__ float fast_tanh(float x) {
    const float e = __expf(-2.0f * fabsf(x));
    const float t = (1.0f - e) * __builtin_amdgcn_rcpf(1.0f + e);
    return copysignf(t, x);
}

// =======================================================================
// Kernel 1: fused forward.  Per wave: 16 rows.  GEMM X(16x128)@W(128x256)
// via v_wmma_f32_16x16x32_bf16, tanh into LDS, then small second layers.
// Outputs per row: v0, delta = r + g*v1 - v0, slp = sum_a(pi_lp - beta_lp).
// =======================================================================
__global__ __launch_bounds__(WAVES * 32) void k_forward(
    const float* __restrict__ state, const float* __restrict__ nstate,
    const float* __restrict__ action, const float* __restrict__ blp,
    const float* __restrict__ reward,
    const float* __restrict__ W1,  const float* __restrict__ b1,
    const float* __restrict__ Wmu, const float* __restrict__ bmu,
    const float* __restrict__ Wlv, const float* __restrict__ blv,
    const float* __restrict__ Wc1, const float* __restrict__ bc1,
    const float* __restrict__ Wc2, const float* __restrict__ bc2,
    float* __restrict__ v0_out, float* __restrict__ delta_out,
    float* __restrict__ slp_out)
{
    __shared__ __bf16 sW[H_N * KP];          // weights, transposed [n][k], bf16 (67.6 KB)
    __shared__ float  sH[WAVES][16][257];    // per-wave tanh(h) strip        (65.8 KB)
    __shared__ float  sB[H_N];               // first-layer bias
    __shared__ float  sWc2[H_N];
    __shared__ float  sWmu[H_N * A_N];
    __shared__ float  sWlv[H_N * A_N];

    const int tid  = threadIdx.x;
    const int lane = tid & 31;
    const int w    = tid >> 5;
    const int lo   = lane & 15;              // col / row-in-strip selector
    const int hi   = lane >> 4;              // K-half / row-half selector
    const size_t rowBase = (size_t)blockIdx.x * ROWS_PER_BLK + (size_t)w * 16;

    // -- cooperative weight load: W(128x256 row-major) -> sW[n*KP + k] bf16 --
    auto loadW = [&](const float* __restrict__ Wg, const float* __restrict__ bg) {
        for (int idx = tid; idx < D_N * H_N; idx += WAVES * 32) {
            int k = idx >> 8, n = idx & 255;
            sW[n * KP + k] = (__bf16)Wg[idx];
        }
        for (int idx = tid; idx < H_N; idx += WAVES * 32) sB[idx] = bg[idx];
    };

    // -- A fragment: 16-bit 16x32 layout. lane: M=lo, K-half = hi*8 (+16). --
    auto loadA = [&](const float* __restrict__ X, v16bf* afr) {
        const float* xr = X + (rowBase + lo) * D_N;
        #pragma unroll
        for (int ks = 0; ks < 4; ++ks) {
            const int kb = ks * 32 + hi * 8;
            v16bf a;
            #pragma unroll
            for (int j = 0; j < 8; ++j) {
                a[j]     = (__bf16)xr[kb + j];
                a[8 + j] = (__bf16)xr[kb + 16 + j];
            }
            afr[ks] = a;
        }
    };

    // -- 16x256 GEMM + bias + tanh -> sH[w].  B frag: lane n=lo, K-half = hi*8 (+16). --
    auto runGemm = [&](const v16bf* afr) {
        for (int nt = 0; nt < 16; ++nt) {
            v8f c = {};
            #pragma unroll
            for (int ks = 0; ks < 4; ++ks) {
                const __bf16* wp = &sW[(nt * 16 + lo) * KP + ks * 32 + hi * 8];
                v16bf b;
                #pragma unroll
                for (int q = 0; q < 8; ++q) { b[q] = wp[q]; b[8 + q] = wp[16 + q]; }
                c = __builtin_amdgcn_wmma_f32_16x16x32_bf16(
                        false, afr[ks], false, b, (short)0, c, false, false);
            }
            const float bias = sB[nt * 16 + lo];
            #pragma unroll
            for (int i = 0; i < 8; ++i)
                sH[w][i + 8 * hi][nt * 16 + lo] = fast_tanh(c[i] + bias);
        }
    };

    // -- second layer: H->1 dot (each lane does half of row `lo`) --
    auto criticDot = [&]() -> float {
        const float* hr = sH[w][lo];
        const int h0 = hi * 128;
        float p = 0.f;
        for (int h = 0; h < 128; ++h) p = fmaf(hr[h0 + h], sWc2[h0 + h], p);
        p += __shfl_xor(p, 16, 32);
        return p + bc2[0];
    };

    // ---------------- phase A: critic(state) ----------------
    loadW(Wc1, bc1);
    for (int idx = tid; idx < H_N; idx += WAVES * 32) sWc2[idx] = Wc2[idx];
    __syncthreads();

    v16bf aS[4];
    loadA(state, aS);
    runGemm(aS);
    __syncthreads();
    const float v0r = criticDot();
    __syncthreads();

    // ---------------- phase B: critic(next_state) ----------------
    v16bf aN[4];
    loadA(nstate, aN);
    runGemm(aN);
    __syncthreads();
    const float v1r = criticDot();
    if (hi == 0) {
        const size_t r = rowBase + lo;
        v0_out[r]    = v0r;
        delta_out[r] = reward[r] + GAMMA_F * v1r - v0r;
    }
    __syncthreads();

    // ---------------- phase C: actor(state) ----------------
    loadW(W1, b1);
    for (int idx = tid; idx < H_N * A_N; idx += WAVES * 32) {
        sWmu[idx] = Wmu[idx];
        sWlv[idx] = Wlv[idx];
    }
    __syncthreads();
    runGemm(aS);               // reuse state fragments
    __syncthreads();

    {
        const float* hr = sH[w][lo];
        const int h0 = hi * 128;
        float pmu[A_N], plv[A_N];
        #pragma unroll
        for (int a = 0; a < A_N; ++a) { pmu[a] = 0.f; plv[a] = 0.f; }
        for (int h = 0; h < 128; ++h) {
            const float hv = hr[h0 + h];
            const int base = (h0 + h) * A_N;
            #pragma unroll
            for (int a = 0; a < A_N; ++a) {
                pmu[a] = fmaf(hv, sWmu[base + a], pmu[a]);
                plv[a] = fmaf(hv, sWlv[base + a], plv[a]);
            }
        }
        #pragma unroll
        for (int a = 0; a < A_N; ++a) {
            pmu[a] += __shfl_xor(pmu[a], 16, 32);
            plv[a] += __shfl_xor(plv[a], 16, 32);
        }
        if (hi == 0) {
            const size_t r = rowBase + lo;
            float slp = 0.f;
            #pragma unroll
            for (int a = 0; a < A_N; ++a) {
                const float mu = pmu[a] + bmu[a];
                const float lv = plv[a] + blv[a];
                const float dmu = action[r * A_N + a] - mu;
                const float lp  = -0.5f * dmu * dmu * __expf(-lv) - 0.5f * lv - HALF_L2PI;
                slp += lp - blp[r * A_N + a];
            }
            slp_out[r] = slp;
        }
    }
}

// =======================================================================
// GAE scan (ret_t = sum_{k>=t} c^{k-t} delta_k), 3-level blocked scan.
// =======================================================================
__global__ __launch_bounds__(32) void k_scan_agg(const float* __restrict__ delta,
                                                 float* __restrict__ Ablk)
{
    const int j = blockIdx.x, t = threadIdx.x;
    const float* d = delta + (size_t)j * SCAN_L + t * 16;
    float s = 0.f;
    #pragma unroll
    for (int e = 15; e >= 0; --e) s = d[e] + CDEC * s;
    float v = __powf(CDEC, 16.0f * (float)t) * s;
    #pragma unroll
    for (int off = 16; off; off >>= 1) v += __shfl_xor(v, off, 32);
    if (t == 0) Ablk[j] = v;
}

__global__ void k_scan_carry(const float* __restrict__ Ablk, float* __restrict__ R)
{
    const float cL = __powf(CDEC, (float)SCAN_L);
    float g = 0.f;
    R[SCAN_NB] = 0.f;
    for (int j = SCAN_NB - 1; j >= 0; --j) {
        g = Ablk[j] + cL * g;
        R[j] = g;
    }
}

__global__ __launch_bounds__(32) void k_scan_apply(
    const float* __restrict__ delta, const float* __restrict__ v0,
    const float* __restrict__ R, float* __restrict__ adv,
    float* __restrict__ psum, float* __restrict__ psq)
{
    __shared__ float sh[32];
    const int j = blockIdx.x, t = threadIdx.x;
    const size_t base = (size_t)j * SCAN_L + t * 16;
    float d[16];
    #pragma unroll
    for (int e = 0; e < 16; ++e) d[e] = delta[base + e];

    float s = 0.f;
    #pragma unroll
    for (int e = 15; e >= 0; --e) s = d[e] + CDEC * s;
    sh[t] = s;
    __syncthreads();

    const float c16 = __powf(CDEC, 16.0f);
    float cc = R[j + 1];
    for (int u = 31; u > t; --u) cc = sh[u] + c16 * cc;   // carry into this chunk

    float g = cc, as = 0.f, aq = 0.f;
    #pragma unroll
    for (int e = 15; e >= 0; --e) {
        g = d[e] + CDEC * g;                 // ret at base+e
        const float a = g - v0[base + e];    // advantage
        adv[base + e] = a;
        as += a; aq += a * a;
    }
    #pragma unroll
    for (int off = 16; off; off >>= 1) {
        as += __shfl_xor(as, off, 32);
        aq += __shfl_xor(aq, off, 32);
    }
    if (t == 0) { psum[j] = as; psq[j] = aq; }
}

// =======================================================================
// Reductions: mean / ddof-1 std / critic loss, then clipped actor loss.
// =======================================================================
__global__ __launch_bounds__(256) void k_stats(const float* __restrict__ psum,
                                               const float* __restrict__ psq,
                                               float* __restrict__ stats)
{
    __shared__ float r1[256], r2[256];
    const int tid = threadIdx.x;
    float s1 = 0.f, s2 = 0.f;
    for (int i = tid; i < SCAN_NB; i += 256) { s1 += psum[i]; s2 += psq[i]; }
    r1[tid] = s1; r2[tid] = s2;
    __syncthreads();
    for (int off = 128; off; off >>= 1) {
        if (tid < off) { r1[tid] += r1[tid + off]; r2[tid] += r2[tid + off]; }
        __syncthreads();
    }
    if (tid == 0) {
        const float S1 = r1[0], S2 = r2[0];
        const float mean = S1 / (float)T_N;
        const float var  = (S2 - S1 * S1 / (float)T_N) / (float)(T_N - 1);
        stats[0] = mean;
        stats[1] = 1.0f / (sqrtf(var) + 1e-7f);
        stats[2] = S2;                       // critic loss
    }
}

__global__ __launch_bounds__(256) void k_actor(const float* __restrict__ adv,
                                               const float* __restrict__ slp,
                                               const float* __restrict__ stats,
                                               float* __restrict__ apart)
{
    __shared__ float red[256];
    const int tid = threadIdx.x;
    const size_t i = (size_t)blockIdx.x * 256 + tid;
    const float mean = stats[0], inv = stats[1];
    const float ah = (adv[i] - mean) * inv;
    const float r  = __expf(slp[i]);
    const float rc = fminf(fmaxf(r, 0.8f), 1.2f);
    red[tid] = fminf(r * ah, rc * ah);
    __syncthreads();
    for (int off = 128; off; off >>= 1) {
        if (tid < off) red[tid] += red[tid + off];
        __syncthreads();
    }
    if (tid == 0) apart[blockIdx.x] = red[0];
}

__global__ __launch_bounds__(256) void k_final(const float* __restrict__ apart,
                                               const float* __restrict__ stats,
                                               float* __restrict__ out)
{
    __shared__ float red[256];
    const int tid = threadIdx.x;
    float s = 0.f;
    for (int i = tid; i < ACT_NB; i += 256) s += apart[i];
    red[tid] = s;
    __syncthreads();
    for (int off = 128; off; off >>= 1) {
        if (tid < off) red[tid] += red[tid + off];
        __syncthreads();
    }
    if (tid == 0) out[0] = stats[2] - red[0];   // critic_loss + (-sum min)
}

// =======================================================================
extern "C" void kernel_launch(void* const* d_in, const int* in_sizes, int n_in,
                              void* d_out, int out_size, void* d_ws, size_t ws_size,
                              hipStream_t stream)
{
    (void)in_sizes; (void)n_in; (void)out_size; (void)ws_size;
    const float* state  = (const float*)d_in[0];
    const float* nstate = (const float*)d_in[1];
    const float* action = (const float*)d_in[2];
    const float* blp    = (const float*)d_in[3];
    const float* reward = (const float*)d_in[4];
    const float* W1  = (const float*)d_in[5];
    const float* b1  = (const float*)d_in[6];
    const float* Wmu = (const float*)d_in[7];
    const float* bmu = (const float*)d_in[8];
    const float* Wlv = (const float*)d_in[9];
    const float* blv = (const float*)d_in[10];
    const float* Wc1 = (const float*)d_in[11];
    const float* bc1 = (const float*)d_in[12];
    const float* Wc2 = (const float*)d_in[13];
    const float* bc2 = (const float*)d_in[14];

    float* ws    = (float*)d_ws;
    float* v0    = ws;
    float* delta = ws + (size_t)T_N;
    float* slp   = ws + 2 * (size_t)T_N;
    float* adv   = ws + 3 * (size_t)T_N;
    float* Ablk  = ws + 4 * (size_t)T_N;
    float* R     = Ablk + SCAN_NB;           // SCAN_NB+1 entries
    float* psum  = R + SCAN_NB + 8;
    float* psq   = psum + SCAN_NB;
    float* stats = psq + SCAN_NB;            // 3 floats (+pad)
    float* apart = stats + 16;               // ACT_NB floats

    k_forward<<<T_N / ROWS_PER_BLK, WAVES * 32, 0, stream>>>(
        state, nstate, action, blp, reward,
        W1, b1, Wmu, bmu, Wlv, blv, Wc1, bc1, Wc2, bc2,
        v0, delta, slp);

    k_scan_agg  <<<SCAN_NB, 32, 0, stream>>>(delta, Ablk);
    k_scan_carry<<<1, 1, 0, stream>>>(Ablk, R);
    k_scan_apply<<<SCAN_NB, 32, 0, stream>>>(delta, v0, R, adv, psum, psq);
    k_stats     <<<1, 256, 0, stream>>>(psum, psq, stats);
    k_actor     <<<ACT_NB, 256, 0, stream>>>(adv, slp, stats, apart);
    k_final     <<<1, 256, 0, stream>>>(apart, stats, (float*)d_out);
}
